// MultiHeadAttention_47132971107268
// MI455X (gfx1250) — matrix operand order repaired and checked by reading, not compiled, not run on hardware
//
#include <hip/hip_runtime.h>

// Problem constants (fixed by the reference)
#define BATCH  2
#define QLEN   2048
#define KLEN   2048
#define DMODEL 1024
#define NHEAD  16
#define DK     64

typedef __attribute__((ext_vector_type(16))) __bf16 v16bf;
typedef __attribute__((ext_vector_type(8)))  __bf16 v8bf;
typedef __attribute__((ext_vector_type(8)))  float  v8f;
typedef int vi4 __attribute__((vector_size(16)));   // 128-bit payload type for async-LDS builtin

#define WMMA_BF16(A, B, C) \
  __builtin_amdgcn_wmma_f32_16x16x32_bf16(false, (A), false, (B), (short)0, (C), false, false)

// ---- CDNA5 async global->LDS path (guarded; falls back to VGPR bounce) ----
#if defined(__has_builtin)
#if __has_builtin(__builtin_amdgcn_global_load_async_to_lds_b128) && \
    __has_builtin(__builtin_amdgcn_s_wait_asynccnt)
#define HAVE_ASYNC_LDS 1
#endif
#endif
#ifndef HAVE_ASYNC_LDS
#define HAVE_ASYNC_LDS 0
#endif

// Copy 16 bytes global -> LDS (per-lane addresses, ASYNCcnt-tracked if available)
__device__ __forceinline__ void stage16(const unsigned short* __restrict__ g,
                                        unsigned short* __restrict__ l) {
#if HAVE_ASYNC_LDS
  __builtin_amdgcn_global_load_async_to_lds_b128(
      (__attribute__((address_space(1))) vi4*)(void*)(unsigned short*)g,
      (__attribute__((address_space(3))) vi4*)(void*)l,
      0, 0);
#else
  *(v8bf*)l = *(const v8bf*)g;
#endif
}
__device__ __forceinline__ void stage_fence() {
#if HAVE_ASYNC_LDS
  __builtin_amdgcn_s_wait_asynccnt(0);
#endif
}

// fp32 -> bf16 round-to-nearest-even (as u16 so host code never touches __bf16)
__device__ __forceinline__ unsigned short f2bfu(float f) {
  unsigned u = __builtin_bit_cast(unsigned, f);
  unsigned r = u + 0x7FFFu + ((u >> 16) & 1u);
  return (unsigned short)(r >> 16);
}

// 16-bit A-matrix fragment (16x32, MxK), contiguous k order used here (not ISA table order):
//   lane L: row M = L&15, K = 16*(L>>4) .. +15 (16 contiguous halfs)
// caller passes p = &A[row*ld + kstep + 16*half]; 16 contiguous halfs.
__device__ __forceinline__ v16bf ld_frag_a(const unsigned short* p) {
  v8bf lo = *(const v8bf*)(p);
  v8bf hi = *(const v8bf*)(p + 8);
  v16bf r;
#pragma unroll
  for (int i = 0; i < 8; ++i) { r[i] = lo[i]; r[i + 8] = hi[i]; }
  return r;
}

// 16-bit B-matrix fragment (32x16, KxN), contiguous k order used here (not ISA table order):
//   lane L: col N = L&15, K = 16*(L>>4) .. +15 (16 contiguous halfs)
__device__ __forceinline__ v16bf ld_frag_b(const unsigned short* p) {
  v8bf lo = *(const v8bf*)(p);
  v8bf hi = *(const v8bf*)(p + 8);
  v16bf r;
#pragma unroll
  for (int i = 0; i < 8; ++i) { r[i] = lo[i]; r[i + 8] = hi[i]; }
  return r;
}

// ---------------------------------------------------------------------------
// Kernel 1: fp32 -> bf16 conversion (4 elements / thread, b128 load, b64 store)
// ---------------------------------------------------------------------------
__global__ void cvt_f32_to_bf16(const float* __restrict__ in,
                                unsigned short* __restrict__ out, int n4) {
  int i = blockIdx.x * blockDim.x + threadIdx.x;
  if (i < n4) {
    float4 v = ((const float4*)in)[i];
    unsigned long long pk =
        (unsigned long long)f2bfu(v.x) |
        ((unsigned long long)f2bfu(v.y) << 16) |
        ((unsigned long long)f2bfu(v.z) << 32) |
        ((unsigned long long)f2bfu(v.w) << 48);
    ((unsigned long long*)out)[i] = pk;
  }
}

// ---------------------------------------------------------------------------
// Kernel 2: generic bf16 GEMM   C[m,n] = sum_k A[m,k]*W[n,k] + bias[n]
// Block = 4 waves sharing one 64-col W block (staged in LDS), each wave owns a
// 16-row tile. K in steps of 32 (wmma bf16): 4 wmma / wave / k-step.
// mode 0: bf16 out, per-head layout  [b, h, len, dk]    (Q', K')
// mode 1: bf16 out, per-head transposed [b, h, dk, len] (V  -> Vt for P.V)
// mode 2: fp32 out, plain [M, Ncols]                    (final projection)
// ---------------------------------------------------------------------------
__global__ void gemm_bf16_wmma(const unsigned short* __restrict__ A,
                               const unsigned short* __restrict__ W,
                               const float* __restrict__ bias,
                               void* __restrict__ out,
                               int M, int Kd, int Ncols, int mode) {
  __shared__ __align__(16) unsigned short wtile[64 * 32];  // [col][k] 4KB

  const int tid    = threadIdx.x;
  const int lane   = tid & 31;
  const int wid    = blockIdx.x * (blockDim.x >> 5) + (tid >> 5);
  const int mtiles = M >> 4;
  const int mt     = wid % mtiles;          // 16-row tile (varies per wave)
  const int ntb    = wid / mtiles;          // 64-col block (shared by block)
  const int half   = lane >> 4;
  const int l15    = lane & 15;

  const v8f vz = {0.f, 0.f, 0.f, 0.f, 0.f, 0.f, 0.f, 0.f};
  v8f acc[4] = {vz, vz, vz, vz};

  const unsigned short* Arow = A + (size_t)(mt * 16 + l15) * Kd;
  const int colbase = ntb << 6;

  // staging assignment: thread copies 32B of the W chunk (col = tid/2, seg = tid&1)
  const int wc  = tid >> 1;
  const int wsg = (tid & 1) * 16;

  for (int kk = 0; kk < Kd; kk += 32) {
    __syncthreads();  // previous iteration's LDS reads complete
    {
      const unsigned short* src = W + (size_t)(colbase + wc) * Kd + kk + wsg;
      unsigned short*       dst = &wtile[wc * 32 + wsg];
      stage16(src, dst);
      stage16(src + 8, dst + 8);
      stage_fence();
    }
    __syncthreads();  // W chunk visible to all waves

    v16bf a = ld_frag_a(Arow + kk + 16 * half);
#pragma unroll
    for (int t = 0; t < 4; ++t) {
      v16bf b = ld_frag_b(&wtile[(t * 16 + l15) * 32 + 16 * half]);
      acc[t] = WMMA_BF16(a, b, acc[t]);
    }
  }

#pragma unroll
  for (int t = 0; t < 4; ++t) {
    const int gcol = colbase + t * 16 + l15;
    const float bv = bias[gcol];
#pragma unroll
    for (int r = 0; r < 8; ++r) {
      const float v  = acc[t][r] + bv;
      const int   gr = mt * 16 + r + 8 * half;     // global row
      if (mode == 2) {
        ((float*)out)[(size_t)gr * Ncols + gcol] = v;
      } else {
        const int bb = gr >> 11;                   // row / 2048
        const int rr = gr & 2047;
        const int hh = gcol >> 6;                  // col / 64
        const int dd = gcol & 63;
        size_t idx = (mode == 0)
            ? ((size_t)((bb * NHEAD + hh) * QLEN + rr) * DK + dd)
            : ((size_t)((bb * NHEAD + hh) * DK + dd) * QLEN + rr);
        ((unsigned short*)out)[idx] = f2bfu(v);
      }
    }
  }
}

// ---------------------------------------------------------------------------
// Kernel 3: flash-attention. Block = 4 waves on 4 consecutive q-tiles of one
// (batch, head). Per 32-key chunk the block stages K (32x64) and V (64x32)
// into LDS once (shared 4 ways); each wave:
//   S = Q.K^T (4 wmma from LDS), mask+scale, online softmax (shfl_xor),
//   P staged through a per-wave LDS tile into A-fragment layout,
//   ctx += P.V (4 wmma from LDS).
// ---------------------------------------------------------------------------
__global__ void attn_wmma(const unsigned short* __restrict__ Qp,
                          const unsigned short* __restrict__ Kp,
                          const unsigned short* __restrict__ Vt,
                          const int* __restrict__ mask,
                          unsigned short* __restrict__ ctx) {
  __shared__ __align__(16) unsigned short kc[32 * 64];      // [krow][d]  4KB
  __shared__ __align__(16) unsigned short vc[64 * 32];      // [d][k]     4KB
  __shared__ __align__(16) unsigned short pbuf[4][16 * 32]; // per-wave P 4KB

  const int tid  = threadIdx.x;
  const int lane = tid & 31;
  const int wib  = tid >> 5;                        // wave in block (0..3)
  const int wgid = blockIdx.x * 4 + wib;
  const int qt   = wgid & 127;                      // q-tile (QLEN/16 = 128)
  const int h    = (wgid >> 7) & 15;                // head  (shared in block)
  const int b    = wgid >> 11;                      // batch (shared in block)
  const int half = lane >> 4;
  const int l15  = lane & 15;

  const unsigned short* Qbase = Qp + ((size_t)(b * NHEAD + h) * QLEN + qt * 16) * DK;
  const unsigned short* Kbase = Kp + (size_t)(b * NHEAD + h) * KLEN * DK;
  const unsigned short* Vbase = Vt + (size_t)(b * NHEAD + h) * DK * KLEN;

  // Q fragments for d = 0..31 and 32..63, kept for the whole loop
  v16bf aq0 = ld_frag_a(Qbase + l15 * DK + 16 * half);
  v16bf aq1 = ld_frag_a(Qbase + l15 * DK + 32 + 16 * half);

  const v8f vz = {0.f, 0.f, 0.f, 0.f, 0.f, 0.f, 0.f, 0.f};
  v8f acc[4] = {vz, vz, vz, vz};
  float m[8], l[8];
#pragma unroll
  for (int r = 0; r < 8; ++r) { m[r] = -3.0e38f; l[r] = 0.f; }

  // staging assignments (128 threads, 64B each per chunk):
  const int krow = tid >> 2, kseg = (tid & 3) * 16;  // K: [32][64], 128B/row
  const int vrow = tid >> 1, vseg = (tid & 1) * 16;  // V: [64][32],  64B/row

  for (int kk = 0; kk < KLEN; kk += 32) {
    __syncthreads();  // previous chunk fully consumed
    {
      const unsigned short* ks = Kbase + (size_t)(kk + krow) * DK + kseg;
      unsigned short*       kd = &kc[krow * 64 + kseg];
      stage16(ks, kd);
      stage16(ks + 8, kd + 8);
      const unsigned short* vs = Vbase + (size_t)vrow * KLEN + kk + vseg;
      unsigned short*       vd = &vc[vrow * 32 + vseg];
      stage16(vs, vd);
      stage16(vs + 8, vd + 8);
      stage_fence();
    }
    __syncthreads();  // K/V chunk visible to all waves

    // ---- scores: two 16x16 tiles over 32 key columns (B frags from LDS) ----
    v8f s0 = vz, s1 = vz;
    {
      v16bf bk = ld_frag_b(&kc[l15 * 64 + 16 * half]);
      s0 = WMMA_BF16(aq0, bk, s0);
      bk = ld_frag_b(&kc[l15 * 64 + 32 + 16 * half]);
      s0 = WMMA_BF16(aq1, bk, s0);
      bk = ld_frag_b(&kc[(l15 + 16) * 64 + 16 * half]);
      s1 = WMMA_BF16(aq0, bk, s1);
      bk = ld_frag_b(&kc[(l15 + 16) * 64 + 32 + 16 * half]);
      s1 = WMMA_BF16(aq1, bk, s1);
    }

    const int colA = kk + l15;
    const int colB = colA + 16;

    // ---- mask + online softmax (row lives in 16 lanes of one half-wave) ----
#pragma unroll
    for (int r = 0; r < 8; ++r) {
      const int q = qt * 16 + r + 8 * half;
      const int* mq = mask + ((size_t)b * QLEN + q) * KLEN;
      float x0 = s0[r] * 0.125f;                 // 1/sqrt(64)
      float x1 = s1[r] * 0.125f;
      if (mq[colA]) x0 = -1.0e12f;
      if (mq[colB]) x1 = -1.0e12f;

      float tm = fmaxf(x0, x1);
      tm = fmaxf(tm, __shfl_xor(tm, 1));
      tm = fmaxf(tm, __shfl_xor(tm, 2));
      tm = fmaxf(tm, __shfl_xor(tm, 4));
      tm = fmaxf(tm, __shfl_xor(tm, 8));

      const float mn = fmaxf(m[r], tm);
      const float sc = __expf(m[r] - mn);
      const float p0 = __expf(x0 - mn);
      const float p1 = __expf(x1 - mn);
      float rs = p0 + p1;
      rs += __shfl_xor(rs, 1);
      rs += __shfl_xor(rs, 2);
      rs += __shfl_xor(rs, 4);
      rs += __shfl_xor(rs, 8);

      l[r] = l[r] * sc + rs;
      m[r] = mn;
      acc[0][r] *= sc; acc[1][r] *= sc; acc[2][r] *= sc; acc[3][r] *= sc;

      // stage P (bf16) row-major 16x32 in this wave's private LDS tile
      pbuf[wib][(r + 8 * half) * 32 + l15]      = f2bfu(p0);
      pbuf[wib][(r + 8 * half) * 32 + l15 + 16] = f2bfu(p1);
    }

    // ---- ctx += P.V : P as A-fragment from LDS, V fragments from LDS ----
    v16bf pa = ld_frag_a(&pbuf[wib][l15 * 32 + 16 * half]);
#pragma unroll
    for (int t = 0; t < 4; ++t) {
      v16bf bv = ld_frag_b(&vc[(t * 16 + l15) * 32 + 16 * half]);
      acc[t] = WMMA_BF16(pa, bv, acc[t]);
    }
  }

  // ---- epilogue: normalize, store bf16 ctx in concat layout [b, q, h*64+d]
#pragma unroll
  for (int t = 0; t < 4; ++t) {
#pragma unroll
    for (int r = 0; r < 8; ++r) {
      const float o = acc[t][r] / l[r];
      const int   q = qt * 16 + r + 8 * half;
      const int   d = h * DK + t * 16 + l15;
      ctx[(size_t)(b * QLEN + q) * DMODEL + d] = f2bfu(o);
    }
  }
}

// ---------------------------------------------------------------------------
// Launch
// ---------------------------------------------------------------------------
extern "C" void kernel_launch(void* const* d_in, const int* in_sizes, int n_in,
                              void* d_out, int out_size, void* d_ws, size_t ws_size,
                              hipStream_t stream) {
  const float* q    = (const float*)d_in[0];
  const float* k    = (const float*)d_in[1];
  const float* v    = (const float*)d_in[2];
  const int*   mask = (const int*)d_in[3];
  const float* Wq   = (const float*)d_in[4];
  const float* bq   = (const float*)d_in[5];
  const float* Wk   = (const float*)d_in[6];
  const float* bk   = (const float*)d_in[7];
  const float* Wv   = (const float*)d_in[8];
  const float* bv   = (const float*)d_in[9];
  const float* Wo   = (const float*)d_in[10];
  const float* bo   = (const float*)d_in[11];

  const size_t SZ_ACT = (size_t)BATCH * QLEN * DMODEL;   // 4,194,304 elems
  const size_t SZ_W   = (size_t)DMODEL * DMODEL;         // 1,048,576 elems

  unsigned short* ws16 = (unsigned short*)d_ws;
  unsigned short* qbf = ws16;
  unsigned short* kbf = qbf + SZ_ACT;
  unsigned short* vbf = kbf + SZ_ACT;
  unsigned short* wqb = vbf + SZ_ACT;
  unsigned short* wkb = wqb + SZ_W;
  unsigned short* wvb = wkb + SZ_W;
  unsigned short* wob = wvb + SZ_W;
  unsigned short* Qp  = wob + SZ_W;     // [b,h,qlen,dk] bf16
  unsigned short* Kp  = Qp + SZ_ACT;    // [b,h,klen,dk] bf16
  unsigned short* Vt  = Kp + SZ_ACT;    // [b,h,dk,klen] bf16 (transposed)
  unsigned short* ctx = Vt + SZ_ACT;    // [b,qlen,dmodel] bf16

  // ---- fp32 -> bf16 conversions ----
  {
    const int na4 = (int)(SZ_ACT / 4), nw4 = (int)(SZ_W / 4);
    cvt_f32_to_bf16<<<(na4 + 255) / 256, 256, 0, stream>>>(q, qbf, na4);
    cvt_f32_to_bf16<<<(na4 + 255) / 256, 256, 0, stream>>>(k, kbf, na4);
    cvt_f32_to_bf16<<<(na4 + 255) / 256, 256, 0, stream>>>(v, vbf, na4);
    cvt_f32_to_bf16<<<(nw4 + 255) / 256, 256, 0, stream>>>(Wq, wqb, nw4);
    cvt_f32_to_bf16<<<(nw4 + 255) / 256, 256, 0, stream>>>(Wk, wkb, nw4);
    cvt_f32_to_bf16<<<(nw4 + 255) / 256, 256, 0, stream>>>(Wv, wvb, nw4);
    cvt_f32_to_bf16<<<(nw4 + 255) / 256, 256, 0, stream>>>(Wo, wob, nw4);
  }

  // ---- GEMM grids: (M/16)*(N/64) waves, 4 waves per 128-thread block ----
  const int M = BATCH * QLEN;                            // 4096
  const int gemmBlocks = (M / 16) * (DMODEL / 64) / 4;   // 1024

  gemm_bf16_wmma<<<gemmBlocks, 128, 0, stream>>>(qbf, wqb, bq, Qp, M, DMODEL, DMODEL, 0);
  gemm_bf16_wmma<<<gemmBlocks, 128, 0, stream>>>(kbf, wkb, bk, Kp, M, DMODEL, DMODEL, 0);
  gemm_bf16_wmma<<<gemmBlocks, 128, 0, stream>>>(vbf, wvb, bv, Vt, M, DMODEL, DMODEL, 1);

  // ---- attention: B*H*(QLEN/16) = 4096 waves -> 1024 blocks of 128 ----
  attn_wmma<<<1024, 128, 0, stream>>>(Qp, Kp, Vt, mask, ctx);

  // ---- output projection: fp32 result straight into d_out ----
  gemm_bf16_wmma<<<gemmBlocks, 128, 0, stream>>>(ctx, wob, bo, d_out, M, DMODEL, DMODEL, 2);
}
